// Attention_37812892074247
// MI455X (gfx1250) — matrix operand order repaired and checked by reading, not compiled, not run on hardware
//
#include <hip/hip_runtime.h>
#include <stdint.h>

typedef __attribute__((ext_vector_type(16))) __bf16 v16bf;
typedef __attribute__((ext_vector_type(8)))  __bf16 v8bf;
typedef __attribute__((ext_vector_type(8)))  float  v8f;

namespace {
constexpr int B_ = 2, T_ = 2048, D_ = 2048, H_ = 16, G_ = 4, HD_ = 128;
constexpr int E_ = H_ * HD_ + 2 * G_ * HD_;           // 3072
constexpr int BT_ = B_ * T_;                          // 4096
constexpr float SCALE_ = 0.08838834764831845f;
}

__device__ __forceinline__ unsigned short f32_to_bf16(float f) {
  union { float f; unsigned int u; } v; v.f = f;
  unsigned int u = v.u;
  return (unsigned short)((u + 0x7FFFu + ((u >> 16) & 1u)) >> 16);
}

__device__ __forceinline__ v8f wmma_bf16(v16bf a, v16bf b, v8f c) {
  return __builtin_amdgcn_wmma_f32_16x16x32_bf16(false, a, false, b, (short)0, c,
                                                 false, false);
}

// A-fragment: 16(M) x 32(K) bf16 tile from row-major [row][K] (stride ld elems).
// Lane l<16: row M=l, elems 0..7 = K h*8.. , elems 8..15 = K 16+h*8.. (ISA 7.12.2)
__device__ __forceinline__ v16bf ld_afrag(const unsigned short* src, int ld) {
  const int lane = threadIdx.x & 31;
  const int half = lane >> 4, l = lane & 15;
  const v8bf lo = *(const v8bf*)(src + (size_t)l * ld + half * 8);
  const v8bf hi = *(const v8bf*)(src + (size_t)l * ld + 16 + half * 8);
  return __builtin_shufflevector(lo, hi, 0,1,2,3,4,5,6,7,8,9,10,11,12,13,14,15);
}

// B-fragment: 32(K) x 16(N) bf16 from row-major [N][K] (stride ld elems).
// Lane l<16: col N=l, elems 0..7 = K h*8.. , elems 8..15 = K 16+h*8.. (matches A-fragment)
__device__ __forceinline__ v16bf ld_bfrag(const unsigned short* src, int ld) {
  const int lane = threadIdx.x & 31;
  const int half = lane >> 4, l = lane & 15;
  const v8bf lo = *(const v8bf*)(src + (size_t)l * ld + half * 8);
  const v8bf hi = *(const v8bf*)(src + (size_t)l * ld + 16 + half * 8);
  return __builtin_shufflevector(lo, hi, 0,1,2,3,4,5,6,7,8,9,10,11,12,13,14,15);
}

// ---------------------------------------------------------------- convert f32->bf16
__global__ void cvt_bf16(const float* __restrict__ in, unsigned short* __restrict__ out, int n) {
  int i = blockIdx.x * blockDim.x + threadIdx.x;
  if (i < n) out[i] = f32_to_bf16(in[i]);
}

// ---------------------------------------------------------------- WMMA GEMM
// C[M][N] = A[M][K] * Bm[N][K]^T ; 64x64 tile per wave
__global__ __launch_bounds__(128) void gemm_bf16_wmma(
    const unsigned short* __restrict__ A, const unsigned short* __restrict__ Bm,
    float* __restrict__ C, int M, int N, int K) {
  const int lane = threadIdx.x & 31;
  const int half = lane >> 4, l = lane & 15;
  const int wave = blockIdx.x * (blockDim.x >> 5) + (threadIdx.x >> 5);
  const int ntn = N >> 6;
  const int tm = (wave / ntn) << 6;
  const int tn = (wave % ntn) << 6;
  if (tm >= M) return;

  v8f acc[4][4];
#pragma unroll
  for (int i = 0; i < 4; i++)
#pragma unroll
    for (int j = 0; j < 4; j++) {
      v8f z = {0.f, 0.f, 0.f, 0.f, 0.f, 0.f, 0.f, 0.f};
      acc[i][j] = z;
    }

  for (int k0 = 0; k0 < K; k0 += 32) {
    if (k0 + 128 < K) {
      __builtin_prefetch(A + (size_t)(tm + lane) * K + k0 + 128, 0, 1);
      __builtin_prefetch(Bm + (size_t)(tn + lane) * K + k0 + 128, 0, 1);
    }
    v16bf a[4], b[4];
#pragma unroll
    for (int i = 0; i < 4; i++) a[i] = ld_afrag(A + (size_t)(tm + i * 16) * K + k0, K);
#pragma unroll
    for (int j = 0; j < 4; j++) b[j] = ld_bfrag(Bm + (size_t)(tn + j * 16) * K + k0, K);
#pragma unroll
    for (int i = 0; i < 4; i++)
#pragma unroll
      for (int j = 0; j < 4; j++) acc[i][j] = wmma_bf16(a[i], b[j], acc[i][j]);
  }
#pragma unroll
  for (int i = 0; i < 4; i++)
#pragma unroll
    for (int j = 0; j < 4; j++)
#pragma unroll
      for (int r = 0; r < 8; r++) {
        const int row = tm + i * 16 + r + half * 8;  // C layout: VGPR r -> M=r(+8)
        const int col = tn + j * 16 + l;
        C[(size_t)row * N + col] = acc[i][j][r];
      }
}

// ---------------------------------------------------------------- qk-norm + RoPE + scatter
// one wave per (b,t,slot) ; slot 0..15 = q head, 16..19 = k head, 20..23 = v head
__global__ __launch_bounds__(256) void qkv_post(
    const float* __restrict__ qkv, unsigned short* __restrict__ qb,
    unsigned short* __restrict__ kb, unsigned short* __restrict__ vtb,
    const int* __restrict__ use_qk_norm) {
  const int wave = blockIdx.x * (blockDim.x >> 5) + (threadIdx.x >> 5);
  if (wave >= BT_ * 24) return;
  const int slot = wave % 24;
  const int bt = wave / 24;
  const int b = bt / T_, t = bt % T_;
  const int lane = threadIdx.x & 31;
  const float* src = qkv + (size_t)bt * E_ + slot * HD_;

  if (slot < H_ + G_) {  // q or k: norm + rope
    float x[4];
#pragma unroll
    for (int j = 0; j < 4; j++) x[j] = src[lane * 4 + j];
    if (*use_qk_norm) {
      float ss = x[0]*x[0] + x[1]*x[1] + x[2]*x[2] + x[3]*x[3];
      ss += __shfl_xor(ss, 1);  ss += __shfl_xor(ss, 2);
      ss += __shfl_xor(ss, 4);  ss += __shfl_xor(ss, 8);
      ss += __shfl_xor(ss, 16);
      const float inv = 1.0f / fmaxf(sqrtf(ss), 1e-10f);
#pragma unroll
      for (int j = 0; j < 4; j++) x[j] *= inv;
    }
    float out[4];
    const float pos = (float)t;
#pragma unroll
    for (int p = 0; p < 2; p++) {
      const int d0 = lane * 4 + 2 * p;
      float se, ce, so, co;
      __sincosf(pos * __powf(10000.f, -(float)(d0 & 63) * (1.f / 64.f)), &se, &ce);
      __sincosf(pos * __powf(10000.f, -(float)((d0 + 1) & 63) * (1.f / 64.f)), &so, &co);
      const float xe = x[2 * p], xo = x[2 * p + 1];
      out[2 * p]     = xe * ce - xo * se;
      out[2 * p + 1] = xo * co + xe * so;
    }
    unsigned short* dst =
        (slot < H_)
            ? qb + (((size_t)b * H_ + slot) * T_ + t) * HD_ + lane * 4
            : kb + (((size_t)b * G_ + (slot - H_)) * T_ + t) * HD_ + lane * 4;
#pragma unroll
    for (int j = 0; j < 4; j++) dst[j] = f32_to_bf16(out[j]);
  } else {  // v: transpose-store [b][g][hd][t]
    const int g = slot - H_ - G_;
    unsigned short* dst = vtb + ((size_t)b * G_ + g) * HD_ * T_;
#pragma unroll
    for (int j = 0; j < 4; j++) {
      const int hd = lane * 4 + j;
      dst[(size_t)hd * T_ + t] = f32_to_bf16(src[hd]);
    }
  }
}

// ---------------------------------------------------------------- flash attention
// one wave per 16-query tile; online softmax; P reshaped via per-wave LDS slab
__global__ __launch_bounds__(128) void attn_wmma(
    const unsigned short* __restrict__ qb, const unsigned short* __restrict__ kb,
    const unsigned short* __restrict__ vtb, unsigned short* __restrict__ ob) {
  __shared__ unsigned short pshare[4][16 * 32];
  const int wv = threadIdx.x >> 5;
  const int lane = threadIdx.x & 31;
  const int half = lane >> 4, l = lane & 15;
  const int gw = blockIdx.x * 4 + wv;
  if (gw >= B_ * H_ * (T_ / 16)) return;
  const int qt = gw & (T_ / 16 - 1);
  const int bh = gw >> 7;
  const int h = bh & (H_ - 1), b = bh >> 4;
  const int g = h / (H_ / G_);

  const unsigned short* qbase = qb + (((size_t)b * H_ + h) * T_ + qt * 16) * HD_;
  const unsigned short* kbase = kb + ((size_t)b * G_ + g) * T_ * HD_;
  const unsigned short* vbase = vtb + ((size_t)b * G_ + g) * HD_ * T_;

  v16bf qf[4];
#pragma unroll
  for (int c = 0; c < 4; c++) qf[c] = ld_afrag(qbase + c * 32, HD_);

  v8f acc[8];
  float m[8], lsum[8];
#pragma unroll
  for (int n = 0; n < 8; n++) {
    v8f z = {0.f, 0.f, 0.f, 0.f, 0.f, 0.f, 0.f, 0.f};
    acc[n] = z;
  }
#pragma unroll
  for (int r = 0; r < 8; r++) { m[r] = -3.0e38f; lsum[r] = 0.f; }

  for (int kt = 0; kt < T_; kt += 32) {
    v8f s0 = {0.f,0.f,0.f,0.f,0.f,0.f,0.f,0.f};
    v8f s1 = {0.f,0.f,0.f,0.f,0.f,0.f,0.f,0.f};
#pragma unroll
    for (int c = 0; c < 4; c++) {
      v16bf b0 = ld_bfrag(kbase + (size_t)kt * HD_ + c * 32, HD_);
      v16bf b1 = ld_bfrag(kbase + (size_t)(kt + 16) * HD_ + c * 32, HD_);
      s0 = wmma_bf16(qf[c], b0, s0);
      s1 = wmma_bf16(qf[c], b1, s1);
    }
    float p0[8], p1[8], alpha[8];
#pragma unroll
    for (int r = 0; r < 8; r++) {
      p0[r] = s0[r] * SCALE_;
      p1[r] = s1[r] * SCALE_;
      float mx = fmaxf(p0[r], p1[r]);             // row r+half*8, reduce 16-lane half
      mx = fmaxf(mx, __shfl_xor(mx, 1));
      mx = fmaxf(mx, __shfl_xor(mx, 2));
      mx = fmaxf(mx, __shfl_xor(mx, 4));
      mx = fmaxf(mx, __shfl_xor(mx, 8));
      const float nm = fmaxf(m[r], mx);
      alpha[r] = __expf(m[r] - nm);
      m[r] = nm;
    }
#pragma unroll
    for (int r = 0; r < 8; r++) {
      p0[r] = __expf(p0[r] - m[r]);
      p1[r] = __expf(p1[r] - m[r]);
      float s = p0[r] + p1[r];
      s += __shfl_xor(s, 1);  s += __shfl_xor(s, 2);
      s += __shfl_xor(s, 4);  s += __shfl_xor(s, 8);
      lsum[r] = lsum[r] * alpha[r] + s;
    }
#pragma unroll
    for (int n = 0; n < 8; n++)
#pragma unroll
      for (int r = 0; r < 8; r++) acc[n][r] *= alpha[r];

    // C-layout -> A-layout via wave-private LDS (row-major 16x32 bf16)
#pragma unroll
    for (int r = 0; r < 8; r++) {
      const int rr = r + half * 8;
      pshare[wv][rr * 32 + l]      = f32_to_bf16(p0[r]);
      pshare[wv][rr * 32 + 16 + l] = f32_to_bf16(p1[r]);
    }
    asm volatile("s_wait_dscnt 0x0" ::: "memory");
    const v8bf plo = *(const v8bf*)&pshare[wv][l * 32 + half * 8];
    const v8bf phi = *(const v8bf*)&pshare[wv][l * 32 + 16 + half * 8];
    const v16bf pa =
        __builtin_shufflevector(plo, phi, 0,1,2,3,4,5,6,7,8,9,10,11,12,13,14,15);

#pragma unroll
    for (int n = 0; n < 8; n++) {
      v16bf vf = ld_bfrag(vbase + (size_t)(n * 16) * T_ + kt, T_);
      acc[n] = wmma_bf16(pa, vf, acc[n]);
    }
  }

#pragma unroll
  for (int r = 0; r < 8; r++) {
    const float inv = 1.0f / lsum[r];
    const int trow = qt * 16 + r + half * 8;
#pragma unroll
    for (int n = 0; n < 8; n++) {
      const int col = h * HD_ + n * 16 + l;
      ob[((size_t)b * T_ + trow) * D_ + col] = f32_to_bf16(acc[n][r] * inv);
    }
  }
}

// ---------------------------------------------------------------- host launch
extern "C" void kernel_launch(void* const* d_in, const int* in_sizes, int n_in,
                              void* d_out, int out_size, void* d_ws, size_t ws_size,
                              hipStream_t stream) {
  (void)in_sizes; (void)n_in; (void)out_size; (void)ws_size;
  const float* x      = (const float*)d_in[0];
  const float* w_qkv  = (const float*)d_in[1];
  const float* w_o    = (const float*)d_in[2];
  const int*   uqn    = (const int*)d_in[4];
  float* out = (float*)d_out;

  char* ws = (char*)d_ws;
  size_t off = 0;
  unsigned short* x_bf    = (unsigned short*)(ws + off); off += (size_t)BT_ * D_ * 2;     // 16.8MB
  unsigned short* wqkv_bf = (unsigned short*)(ws + off); off += (size_t)E_ * D_ * 2;      // 12.6MB
  unsigned short* wo_bf   = (unsigned short*)(ws + off); off += (size_t)D_ * D_ * 2;      // 8.4MB
  float*          qkv_f   = (float*)(ws + off);          off += (size_t)BT_ * E_ * 4;     // 50.3MB
  unsigned short* q_bf    = (unsigned short*)(ws + off); off += (size_t)B_*H_*T_*HD_ * 2; // 16.8MB
  unsigned short* k_bf    = (unsigned short*)(ws + off); off += (size_t)B_*G_*T_*HD_ * 2; // 4.2MB
  unsigned short* vt_bf   = (unsigned short*)(ws + off); off += (size_t)B_*G_*HD_*T_ * 2; // 4.2MB
  unsigned short* ao_bf   = (unsigned short*)(ws + off); off += (size_t)BT_ * D_ * 2;     // 16.8MB

  // 1) down-convert operands to bf16
  {
    int n1 = BT_ * D_, n2 = E_ * D_, n3 = D_ * D_;
    cvt_bf16<<<(n1 + 255) / 256, 256, 0, stream>>>(x, x_bf, n1);
    cvt_bf16<<<(n2 + 255) / 256, 256, 0, stream>>>(w_qkv, wqkv_bf, n2);
    cvt_bf16<<<(n3 + 255) / 256, 256, 0, stream>>>(w_o, wo_bf, n3);
  }
  // 2) qkv = x @ w_qkv^T    (M=4096, N=3072, K=2048); 64x64 waves, 4 waves/block
  {
    int tiles = (BT_ / 64) * (E_ / 64);
    gemm_bf16_wmma<<<(tiles + 3) / 4, 128, 0, stream>>>(x_bf, wqkv_bf, qkv_f, BT_, E_, D_);
  }
  // 3) qk-norm + rope + scatter (v transposed)
  {
    int waves = BT_ * 24;
    qkv_post<<<(waves + 7) / 8, 256, 0, stream>>>(qkv_f, q_bf, k_bf, vt_bf, uqn);
  }
  // 4) flash attention
  {
    int qtiles = B_ * H_ * (T_ / 16);
    attn_wmma<<<(qtiles + 3) / 4, 128, 0, stream>>>(q_bf, k_bf, vt_bf, ao_bf);
  }
  // 5) out = attn @ w_o^T   (M=4096, N=2048, K=2048)
  {
    int tiles = (BT_ / 64) * (D_ / 64);
    gemm_bf16_wmma<<<(tiles + 3) / 4, 128, 0, stream>>>(ao_bf, wo_bf, out, BT_, D_, D_);
  }
}
